// VxNet_62010737819735
// MI455X (gfx1250) — matrix operand order repaired and checked by reading, not compiled, not run on hardware
//
#include <hip/hip_runtime.h>
#include <hip/hip_bf16.h>

// ---------------------------------------------------------------------------
// CDNA5 (gfx1250) sparse-conv VoxelNet.
// spconv = gather-GEMM on v_wmma_f32_16x16x32_f16:
//   - workgroup = (Cout/16) waves sharing one 16-row gathered A tile in LDS
//   - A: double-buffered async global->LDS gather (ASYNCcnt)
//   - B: double-buffered TDM tensor_load_to_lds weight panel (TENSORcnt)
// ---------------------------------------------------------------------------

typedef __attribute__((ext_vector_type(16))) _Float16 v16h;
typedef __attribute__((ext_vector_type(8)))  float    v8f;
typedef __attribute__((ext_vector_type(4)))  unsigned int v4u;
typedef __attribute__((ext_vector_type(8)))  int      v8i;
typedef __attribute__((ext_vector_type(4)))  int      v4i;

#define KT_MAX 27

#if defined(__AMDGCN__) && defined(__gfx1250__) && \
    __has_builtin(__builtin_amdgcn_global_load_async_to_lds_b32) && \
    __has_builtin(__builtin_amdgcn_s_wait_asynccnt)
#define HAS_ASYNC_LDS 1
typedef __attribute__((address_space(1))) int async_gptr_t;  // global src
typedef __attribute__((address_space(3))) int async_lptr_t;  // LDS dst
#else
#define HAS_ASYNC_LDS 0
#endif

#if defined(__AMDGCN__) && defined(__gfx1250__) && \
    __has_builtin(__builtin_amdgcn_tensor_load_to_lds) && \
    __has_builtin(__builtin_amdgcn_s_wait_tensorcnt)
#define HAS_TDM 1
#else
#define HAS_TDM 0
#endif

// --------------------------- f32 -> f16 convert ----------------------------
__global__ __launch_bounds__(256) void f32_to_f16_kernel(
    const float* __restrict__ in, _Float16* __restrict__ out, int n) {
  int i = blockIdx.x * 256 + threadIdx.x;
  if (i < n) out[i] = (_Float16)in[i];
}

#if HAS_TDM
__device__ __forceinline__ unsigned lds_byte_addr(const void* p) {
  return (unsigned)(unsigned long)
      (__attribute__((address_space(3))) const void*)p;
}

// TDM: stage a [32 x Cout] f16 weight tile (row stride Cout) into LDS.
// rows = remaining K rows; reads past tensor_dim1 return zero (K tail).
__device__ __forceinline__ void tdm_stage_wtile(
    const _Float16* wrow0, int rows, int Cout, _Float16* wbuf) {
  unsigned long ga = (unsigned long)(const void*)wrow0;
  unsigned lb = lds_byte_addr(wbuf);
  v4u g0;
  g0[0] = 1u;                                           // count=1, user D#
  g0[1] = lb;                                           // lds_addr (bytes)
  g0[2] = (unsigned)(ga & 0xFFFFFFFFu);                 // global_addr[31:0]
  g0[3] = (unsigned)((ga >> 32) & 0x01FFFFFFu) | (2u << 30);  // [56:32]+type=2
  unsigned ur = (unsigned)rows, uc = (unsigned)Cout;
  v8i g1;
  g1[0] = (int)(1u << 16);                              // data_size=2 bytes
  g1[1] = (int)((uc & 0xFFFFu) << 16);                  // tensor_dim0 lo
  g1[2] = (int)((uc >> 16) | ((ur & 0xFFFFu) << 16));   // td0 hi | td1 lo
  g1[3] = (int)((ur >> 16) | (uc << 16));               // td1 hi | tile_dim0
  g1[4] = 32;                                           // tile_dim1=32, td2=0
  g1[5] = (int)uc;                                      // dim0 stride lo
  g1[6] = 0;                                            // stride hi | d1str lo
  g1[7] = 0;
  v4i g2 = {0, 0, 0, 0};
  v4i g3 = {0, 0, 0, 0};
#if __clang_major__ >= 23
  v8i g4 = {0, 0, 0, 0, 0, 0, 0, 0};
  __builtin_amdgcn_tensor_load_to_lds(g0, g1, g2, g3, g4, 0);
#else
  __builtin_amdgcn_tensor_load_to_lds(g0, g1, g2, g3, 0);
#endif
}
#endif

// ---------------- per-chunk gather stage: global -> LDS (f32) --------------
__device__ __forceinline__ void stage_chunk(
    const float* __restrict__ fin, const int* __restrict__ rb,
    const int* lds_rb, int row0, int Nout, int Ktaps, int Cin, int tot,
    int kk, float* gbuf, int tid, int nthr)
{
  for (int t = tid; t < 16 * 32; t += nthr) {
    int r = t >> 5;            // row in tile
    int e = kk + (t & 31);     // flattened K element (tap*Cin + c)
    int  idx = -1;
    long srcoff = 0;
    if (e < tot) {
      int tap = e / Cin;
      int c   = e - tap * Cin;
      if (rb) idx = lds_rb[r * Ktaps + tap];
      else    { int row = row0 + r; idx = (row < Nout) ? row : -1; }
      srcoff = (long)idx * Cin + c;
    }
    if (idx >= 0) {
#if HAS_ASYNC_LDS
      __builtin_amdgcn_global_load_async_to_lds_b32(
          (async_gptr_t*)(fin + srcoff),
          (async_lptr_t*)(gbuf + t),
          /*offset=*/0, /*cpol=*/0);
#else
      gbuf[t] = fin[srcoff];
#endif
    } else {
      gbuf[t] = 0.0f;
    }
  }
}

// ------------------------- gather-GEMM via WMMA ----------------------------
// D[row, n0+j] = sum_e gather(row, e) * W[e, n0+j],  e = tap*Cin + c
// rb == nullptr -> identity gather (dense GEMM, Ktaps must be 1).
// Block: 32 * (Cout/16) threads; wave w computes output-channel tile w.
__global__ __launch_bounds__(128) void spconv_wmma_kernel(
    const float* __restrict__ fin, const int* __restrict__ rb,
    int Ktaps, int Cin, const _Float16* __restrict__ wgt,
    const float* __restrict__ bn, float* __restrict__ fout,
    int Nout, int Cout, int relu)
{
  __shared__ int lds_rb[16 * KT_MAX];
  __shared__ __align__(16) float gbuf[2][16 * 32];
#if HAS_TDM
  __shared__ __align__(16) _Float16 wbuf[2][32 * 64];   // 32 K-rows x Cout<=64
#endif

  const int tid  = threadIdx.x;
  const int nthr = blockDim.x;
  const int lane = tid & 31;
  const int wave = tid >> 5;
  const int row0 = blockIdx.x * 16;
  const int n0   = wave * 16;
  const int tot  = Ktaps * Cin;

  if (rb) {
    for (int t = tid; t < 16 * Ktaps; t += nthr) {
      int r = t / Ktaps, tap = t - r * Ktaps;
      int row = row0 + r;
      lds_rb[t] = (row < Nout) ? rb[row * Ktaps + tap] : -1;
    }
    __syncthreads();
  }

  v8f acc = {};
  const int ncol = lane & 15;
  const int hi   = lane >> 4;
  // CDNA5 16-bit A (16x32) fragment positions for this lane:
  // lanes 0-15:  row=lane,    K[0..7]  and K[16..23]
  // lanes 16-31: row=lane-16, K[8..15] and K[24..31]
  const int mrow = ncol;
  const int g0i = hi * 8;
  const int g1i = 16 + hi * 8;
  const int kb  = hi * 16;    // B: same K positions as A (g0i.., g1i..)

  // prologue: stage chunk 0 (A gather + B weight tile)
  stage_chunk(fin, rb, lds_rb, row0, Nout, Ktaps, Cin, tot, 0,
              gbuf[0], tid, nthr);
#if HAS_TDM
  if (wave == 0)
    tdm_stage_wtile(wgt, tot, Cout, wbuf[0]);
#endif

  int cur = 0;
  for (int kk = 0; kk < tot; kk += 32) {
#if HAS_ASYNC_LDS
    __builtin_amdgcn_s_wait_asynccnt(0);   // our async gather done
#endif
#if HAS_TDM
    __builtin_amdgcn_s_wait_tensorcnt(0);  // our TDM tile done (wave0)
#endif
    __syncthreads();                       // all waves' stages visible,
                                           // previous-chunk reads retired
    if (kk + 32 < tot) {
      stage_chunk(fin, rb, lds_rb, row0, Nout, Ktaps, Cin, tot, kk + 32,
                  gbuf[cur ^ 1], tid, nthr);
#if HAS_TDM
      if (wave == 0)
        tdm_stage_wtile(wgt + (long)(kk + 32) * Cout, tot - (kk + 32),
                        Cout, wbuf[cur ^ 1]);
#endif
    }

    // ---- A fragment: read fp32 tile from LDS (b128), convert to f16 ----
    const float* gc = gbuf[cur];
    const float4* p0 = reinterpret_cast<const float4*>(&gc[mrow * 32 + g0i]);
    const float4* p1 = reinterpret_cast<const float4*>(&gc[mrow * 32 + g1i]);
    float4 f0 = p0[0], f1 = p0[1], f2 = p1[0], f3 = p1[1];
    v16h a;
    a[0]  = (_Float16)f0.x; a[1]  = (_Float16)f0.y;
    a[2]  = (_Float16)f0.z; a[3]  = (_Float16)f0.w;
    a[4]  = (_Float16)f1.x; a[5]  = (_Float16)f1.y;
    a[6]  = (_Float16)f1.z; a[7]  = (_Float16)f1.w;
    a[8]  = (_Float16)f2.x; a[9]  = (_Float16)f2.y;
    a[10] = (_Float16)f2.z; a[11] = (_Float16)f2.w;
    a[12] = (_Float16)f3.x; a[13] = (_Float16)f3.y;
    a[14] = (_Float16)f3.z; a[15] = (_Float16)f3.w;

    // ---- B fragment: column n0+ncol, K rows kk+g0i..+7 and kk+g1i..+7 ----
    v16h b;
#if HAS_TDM
    const _Float16* wb = wbuf[cur];
    #pragma unroll
    for (int i = 0; i < 16; ++i)
      b[i] = wb[(i < 8 ? g0i + i : g1i + i - 8) * Cout + n0 + ncol];   // OOB K rows were zero-filled
#else
    #pragma unroll
    for (int i = 0; i < 16; ++i) {
      int e = kk + (i < 8 ? g0i + i : g1i + i - 8);
      b[i] = (e < tot) ? wgt[(long)e * Cout + n0 + ncol] : (_Float16)0.0f;
    }
    if (kk + 32 < tot)
      __builtin_prefetch(&wgt[(long)(kk + 32) * Cout + n0], 0, 1);
#endif

    acc = __builtin_amdgcn_wmma_f32_16x16x32_f16(
        /*neg_a=*/false, a, /*neg_b=*/false, b,
        /*c_mod=*/(short)0, acc, /*reuse_a=*/false, /*reuse_b=*/false);
    cur ^= 1;
  }

  // ---- epilogue: fused BN + ReLU, f32 C/D layout (8 VGPRs) ----
  const int col = n0 + ncol;
  float sc = 1.0f, bi = 0.0f, mu = 0.0f;
  if (bn) {
    float g  = bn[0 * Cout + col];
    float be = bn[1 * Cout + col];
    mu       = bn[2 * Cout + col];
    float va = bn[3 * Cout + col];
    sc = g * rsqrtf(va + 1e-3f);
    bi = be;
  }
  #pragma unroll
  for (int j = 0; j < 8; ++j) {
    int m   = j + (hi << 3);        // lanes 0-15: M=j ; lanes 16-31: M=j+8
    int row = row0 + m;
    if (row < Nout) {
      float y = acc[j];
      y = (y - mu) * sc + bi;
      if (relu) y = fmaxf(y, 0.0f);
      fout[(long)row * Cout + col] = y;
    }
  }
}

// --------------------- 3-NN inverse-distance interpolation -----------------
__global__ __launch_bounds__(256) void interp_knn_kernel(
    const float* __restrict__ points_mean,  // Np x 4  (cols 1..3 = xyz)
    const int*   __restrict__ coords,       // Nv x 4  (b,z,y,x)
    int Nv, float vx, float vy, float vz,
    const float* __restrict__ feats, int C,
    float* __restrict__ outp, int outStride, int outOffset, int Np)
{
  __shared__ float sx[256], sy[256], sz[256];
  int i = blockIdx.x * 256 + threadIdx.x;
  float px = 0.f, py = 0.f, pz = 0.f;
  if (i < Np) {
    px = points_mean[i * 4 + 1];
    py = points_mean[i * 4 + 2];
    pz = points_mean[i * 4 + 3];
  }
  float d0 = 3e38f, d1 = 3e38f, d2 = 3e38f;
  int   i0 = 0, i1 = 0, i2 = 0;

  for (int base = 0; base < Nv; base += 256) {
    int v = base + threadIdx.x;
    if (v < Nv) {
      sz[threadIdx.x] = (float)coords[v * 4 + 1] * vz - 3.0f  + 0.5f * vz;
      sy[threadIdx.x] = (float)coords[v * 4 + 2] * vy - 40.0f + 0.5f * vy;
      sx[threadIdx.x] = (float)coords[v * 4 + 3] * vx          + 0.5f * vx;
    }
    __syncthreads();
    int lim = min(256, Nv - base);
    for (int t = 0; t < lim; ++t) {
      float dx = px - sx[t], dy = py - sy[t], dz = pz - sz[t];
      float d  = dx * dx + dy * dy + dz * dz;
      int  vi  = base + t;
      if (d < d2) {
        if (d < d0)      { d2 = d1; i2 = i1; d1 = d0; i1 = i0; d0 = d; i0 = vi; }
        else if (d < d1) { d2 = d1; i2 = i1; d1 = d;  i1 = vi; }
        else             { d2 = d;  i2 = vi; }
      }
    }
    __syncthreads();
  }

  if (i < Np) {
    float w0 = 1.0f / (d0 + 1e-8f);
    float w1 = 1.0f / (d1 + 1e-8f);
    float w2 = 1.0f / (d2 + 1e-8f);
    float ws = w0 + w1 + w2;
    w0 /= ws; w1 /= ws; w2 /= ws;
    for (int c = 0; c < C; ++c) {
      float f = feats[(long)i0 * C + c] * w0 +
                feats[(long)i1 * C + c] * w1 +
                feats[(long)i2 * C + c] * w2;
      outp[(long)i * outStride + outOffset + c] = f;
    }
  }
}

// ----------------------------- cls / reg heads -----------------------------
__global__ __launch_bounds__(256) void heads_kernel(
    const float* __restrict__ pw2,   // Np x 64
    const float* __restrict__ wcls,  // 64 x 1
    const float* __restrict__ wreg,  // 64 x 3
    float* __restrict__ cls_out, float* __restrict__ reg_out, int Np)
{
  int i = blockIdx.x * 256 + threadIdx.x;
  if (i >= Np) return;
  float c = 0.f, r0 = 0.f, r1 = 0.f, r2 = 0.f;
  #pragma unroll 8
  for (int k = 0; k < 64; ++k) {
    float f = pw2[(long)i * 64 + k];
    c  += f * wcls[k];
    r0 += f * wreg[k * 3 + 0];
    r1 += f * wreg[k * 3 + 1];
    r2 += f * wreg[k * 3 + 2];
  }
  cls_out[i] = c;
  reg_out[i * 3 + 0] = r0;
  reg_out[i * 3 + 1] = r1;
  reg_out[i * 3 + 2] = r2;
}

// ---------------------------------------------------------------------------
extern "C" void kernel_launch(void* const* d_in, const int* in_sizes, int n_in,
                              void* d_out, int out_size, void* d_ws, size_t ws_size,
                              hipStream_t stream) {
  (void)n_in; (void)out_size; (void)ws_size;

  const float* x_feats     = (const float*)d_in[0];
  const float* points_mean = (const float*)d_in[1];

  const int widx[15] = {2, 4, 6, 8, 10, 12, 14, 16, 18, 20, 22, 24, 26, 28, 30};
  const int bidx[14] = {3, 5, 7, 9, 11, 13, 15, 17, 19, 21, 23, 25, 27, 29};
  const float* wcls = (const float*)d_in[31];
  const float* wreg = (const float*)d_in[32];

  const int* rb_sub0 = (const int*)d_in[33];
  const int* rb_d0   = (const int*)d_in[34];
  const int* rb_sub1 = (const int*)d_in[35];
  const int* rb_d1   = (const int*)d_in[36];
  const int* rb_sub2 = (const int*)d_in[37];
  const int* rb_d2   = (const int*)d_in[38];
  const int* rb_sub3 = (const int*)d_in[39];
  const int* rb_ex   = (const int*)d_in[40];
  const int* coords1 = (const int*)d_in[41];
  const int* coords2 = (const int*)d_in[42];
  const int* coords3 = (const int*)d_in[43];

  const int N0  = in_sizes[33] / 27;
  const int N1  = in_sizes[34] / 27;
  const int N2  = in_sizes[36] / 27;
  const int N3  = in_sizes[38] / 27;
  const int Nex = in_sizes[40] / 2;
  const int Np  = in_sizes[1] / 4;   // 4096

  // ---------------- workspace partition ----------------
  size_t off = 0;
  auto wsAlloc = [&](size_t bytes) -> char* {
    size_t o = off;
    off = (off + bytes + 255) & ~(size_t)255;
    return (char*)d_ws + o;
  };

  _Float16* w16[15];
  for (int i = 0; i < 15; ++i)
    w16[i] = (_Float16*)wsAlloc((size_t)in_sizes[widx[i]] * sizeof(_Float16));

  size_t bufElems = (size_t)N0 * 16;
  if ((size_t)N1 * 32 > bufElems) bufElems = (size_t)N1 * 32;
  if ((size_t)N2 * 64 > bufElems) bufElems = (size_t)N2 * 64;
  if ((size_t)N3 * 64 > bufElems) bufElems = (size_t)N3 * 64;
  float* bufA = (float*)wsAlloc(bufElems * sizeof(float));
  float* bufB = (float*)wsAlloc(bufElems * sizeof(float));
  float* pw   = (float*)wsAlloc((size_t)Np * 160 * sizeof(float));
  float* pw2  = (float*)wsAlloc((size_t)Np * 64  * sizeof(float));

  // ---------------- convert all weights to f16 ----------------
  for (int i = 0; i < 15; ++i) {
    int n = in_sizes[widx[i]];
    f32_to_f16_kernel<<<(n + 255) / 256, 256, 0, stream>>>(
        (const float*)d_in[widx[i]], w16[i], n);
  }

  auto bn = [&](int layer) { return (const float*)d_in[bidx[layer]]; };

  auto spconv = [&](const float* fin, const int* rb, int Ktaps, int Cin,
                    const _Float16* wgt, const float* bnp, float* fout,
                    int Nout, int Cout, bool relu) {
    dim3 grid((Nout + 15) / 16);
    dim3 block(2 * Cout);            // 32 * (Cout/16) threads
    spconv_wmma_kernel<<<grid, block, 0, stream>>>(
        fin, rb, Ktaps, Cin, wgt, bnp, fout, Nout, Cout, relu ? 1 : 0);
  };

  // ---------------- network ----------------
  spconv(x_feats, rb_sub0, 27,  4, w16[0], bn(0),  bufA, N0, 16, true);  // 0a
  spconv(bufA,    rb_sub0, 27, 16, w16[1], bn(1),  bufB, N0, 16, true);  // 0b
  spconv(bufB,    rb_d0,   27, 16, w16[2], bn(2),  bufA, N1, 32, true);  // d0
  spconv(bufA,    rb_sub1, 27, 32, w16[3], bn(3),  bufB, N1, 32, true);  // 1a
  spconv(bufB,    rb_sub1, 27, 32, w16[4], bn(4),  bufA, N1, 32, true);  // 1b
  interp_knn_kernel<<<(Np + 255) / 256, 256, 0, stream>>>(
      points_mean, coords1, N1, 0.1f, 0.1f, 0.2f, bufA, 32, pw, 160, 0, Np);
  spconv(bufA,    rb_d1,   27, 32, w16[5], bn(5),  bufB, N2, 64, true);  // d1
  spconv(bufB,    rb_sub2, 27, 64, w16[6], bn(6),  bufA, N2, 64, true);  // 2a
  spconv(bufA,    rb_sub2, 27, 64, w16[7], bn(7),  bufB, N2, 64, true);  // 2b
  spconv(bufB,    rb_sub2, 27, 64, w16[8], bn(8),  bufA, N2, 64, true);  // 2c
  interp_knn_kernel<<<(Np + 255) / 256, 256, 0, stream>>>(
      points_mean, coords2, N2, 0.2f, 0.2f, 0.4f, bufA, 64, pw, 160, 32, Np);
  spconv(bufA,    rb_d2,   27, 64, w16[9], bn(9),  bufB, N3, 64, true);  // d2
  spconv(bufB,    rb_sub3, 27, 64, w16[10], bn(10), bufA, N3, 64, true); // 3a
  spconv(bufA,    rb_sub3, 27, 64, w16[11], bn(11), bufB, N3, 64, true); // 3b
  spconv(bufB,    rb_sub3, 27, 64, w16[12], bn(12), bufA, N3, 64, true); // 3c
  interp_knn_kernel<<<(Np + 255) / 256, 256, 0, stream>>>(
      points_mean, coords3, N3, 0.4f, 0.4f, 0.8f, bufA, 64, pw, 160, 96, Np);

  // output sections
  float* out_vox = (float*)d_out;                 // Nex x 64
  float* out_pm  = out_vox + (size_t)Nex * 64;    // Np x 4
  float* out_cls = out_pm + (size_t)Np * 4;       // Np
  float* out_reg = out_cls + (size_t)Np;          // Np x 3

  spconv(bufA, rb_ex, 2, 64, w16[13], bn(13), out_vox, Nex, 64, true);
  spconv(pw, nullptr, 1, 160, w16[14], nullptr, pw2, Np, 64, false);

  heads_kernel<<<(Np + 255) / 256, 256, 0, stream>>>(
      pw2, wcls, wreg, out_cls, out_reg, Np);

  (void)hipMemcpyAsync(out_pm, points_mean, (size_t)Np * 4 * sizeof(float),
                       hipMemcpyDeviceToDevice, stream);
}
